// CombineGraph_9998683865141
// MI455X (gfx1250) — compile-verified, matrix operand order checked by reading, not run on hardware
//
#include <hip/hip_runtime.h>
#include <hip/hip_bf16.h>

// ---------------------------------------------------------------------------
// CombineGraph forward for MI455X (gfx1250, wave32, WMMA).
// B=128, S=50, D=128, NUM_NODE=40000, SAMPLE=12, HOP=2, leaky slope 0.2.
//
// Roofline: embedding table = 40000*128*4B = 20.5MB -> L2 resident (192MB).
// Total GEMM work ~42 GFLOP; dominant GEMMs (feat@w1 K=129, [self|neigh]@w3
// K=256) run on v_wmma_f32_16x16x32_bf16 with f32 accumulation.
// A-matrix fragments are staged once per block in row-major bf16 LDS and
// fetched per-wave as two ds_load_b128 each (ISA 7.12.2 16-bit A layout),
// eliminating redundant per-wave convert chains.
// ---------------------------------------------------------------------------

#define B_   128
#define S_   50
#define D_   128
#define NN_  40000
#define SMP_ 12
#define TL_  8          // session-nodes per workgroup tile in global agg
#define FP_  136        // padded feat row stride (ushorts): 272B -> bank rotate
#define AP_  264        // padded GEMM2 A row stride (ushorts): 528B

typedef __attribute__((ext_vector_type(16))) __bf16 v16bf;
typedef __attribute__((ext_vector_type(8)))  __bf16 v8bf;
typedef __attribute__((ext_vector_type(8)))  float  v8f;

__device__ __forceinline__ unsigned short f2bfbits(float f) {
  unsigned u = __builtin_bit_cast(unsigned, f);
  unsigned r = u + 0x7FFFu + ((u >> 16) & 1u);   // round-to-nearest-even
  return (unsigned short)(r >> 16);
}
__device__ __forceinline__ float bf2f(unsigned short h) {
  return __builtin_bit_cast(float, ((unsigned)h) << 16);
}
__device__ __forceinline__ int clampi(int v) {
  return v < 0 ? 0 : (v >= NN_ ? NN_ - 1 : v);
}
__device__ __forceinline__ float leaky02(float x) { return x >= 0.f ? x : 0.2f * x; }

// K index inside a 32-wide WMMA K-step for 16-bit A/B fragments (ISA 7.12.2):
// lane 0-15 -> k base 0, lane 16-31 -> k base 8; e<8 -> k=base+e, e>=8 -> 16+base+(e-8)
#define KPAT(e, lane) ((((e) >> 3) << 4) + ((((lane) >> 4)) << 3) + ((e) & 7))

// A fragment from a row-major bf16 row in LDS: two contiguous 16B runs.
__device__ __forceinline__ v16bf load_afrag(const unsigned short* row, int kb, int hi8) {
  v8bf lo = *(const v8bf*)(row + kb + hi8);
  v8bf hh = *(const v8bf*)(row + kb + 16 + hi8);
  return __builtin_shufflevector(lo, hh, 0, 1, 2, 3, 4, 5, 6, 7,
                                 8, 9, 10, 11, 12, 13, 14, 15);
}

// ---------------------------------------------------------------------------
// Prep: shuffle w1 ([2,129,128], first 128 K rows) and w3 ([2,256,128]) into
// bf16 WMMA B-fragment order: [hop][ks][nt][lane][e].
// ---------------------------------------------------------------------------
__global__ void prep_frags_kernel(const float* __restrict__ w1,
                                  const float* __restrict__ w3,
                                  unsigned short* __restrict__ w1f,
                                  unsigned short* __restrict__ w3f) {
  int i = blockIdx.x * blockDim.x + threadIdx.x;
  const int N1 = 2 * 4 * 8 * 32 * 16;   // 32768
  const int N3 = 2 * 8 * 8 * 32 * 16;   // 65536
  if (i < N1) {
    int e = i & 15, lane = (i >> 4) & 31, nt = (i >> 9) & 7, ks = (i >> 12) & 3, hop = i >> 14;
    int k = ks * 32 + KPAT(e, lane);
    int n = nt * 16 + (lane & 15);
    w1f[i] = f2bfbits(w1[((hop * 129) + k) * 128 + n]);
  } else if (i < N1 + N3) {
    int j = i - N1;
    int e = j & 15, lane = (j >> 4) & 31, nt = (j >> 9) & 7, ks = (j >> 12) & 7, hop = (j >> 15) & 1;
    int k = ks * 32 + KPAT(e, lane);
    int n = nt * 16 + (lane & 15);
    w3f[j] = f2bfbits(w3[((hop * 256) + k) * 128 + n]);
  }
}

// ---------------------------------------------------------------------------
// Neighbor sampling index expansion.
// ---------------------------------------------------------------------------
__global__ void build_n1_kernel(const int* __restrict__ inputs,
                                const int* __restrict__ adj_all,
                                const float* __restrict__ num_weight,
                                int* __restrict__ n1, float* __restrict__ nw1) {
  int i = blockIdx.x * blockDim.x + threadIdx.x;
  if (i >= B_ * S_ * SMP_) return;
  int b = i / (S_ * SMP_);
  int rem = i - b * (S_ * SMP_);
  int s = rem / SMP_, j = rem - s * SMP_;
  int node = clampi(inputs[b * S_ + s]);
  n1[i]  = adj_all[node * SMP_ + j];
  nw1[i] = num_weight[node * SMP_ + j];
}

__global__ void build_n2_kernel(const int* __restrict__ n1,
                                const int* __restrict__ adj_all,
                                const float* __restrict__ num_weight,
                                int* __restrict__ n2, float* __restrict__ nw2) {
  int i = blockIdx.x * blockDim.x + threadIdx.x;
  const int L1 = S_ * SMP_;                 // 600
  if (i >= B_ * L1 * SMP_) return;
  int b = i / (L1 * SMP_);
  int rem = i - b * (L1 * SMP_);
  int l = rem / SMP_, j = rem - l * SMP_;
  int node = clampi(n1[b * L1 + l]);
  n2[i]  = adj_all[node * SMP_ + j];
  nw2[i] = num_weight[node * SMP_ + j];
}

// ---------------------------------------------------------------------------
// session[b,d] = sum_s emb[item[b,s],d]*mask[b,s] / sum_s mask[b,s]
// ---------------------------------------------------------------------------
__global__ void session_kernel(const float* __restrict__ emb,
                               const int* __restrict__ item,
                               const int* __restrict__ mask,
                               float* __restrict__ session) {
  int b = blockIdx.x, d = threadIdx.x;
  float acc = 0.f, msum = 0.f;
  for (int s = 0; s < S_; ++s) {
    int mk = mask[b * S_ + s];
    msum += (float)mk;
    if (mk) acc += emb[(size_t)clampi(item[b * S_ + s]) * D_ + d];
  }
  session[b * D_ + d] = acc / msum;
}

// ---------------------------------------------------------------------------
// Local attention aggregation -> writes h_local into d_out.
// ---------------------------------------------------------------------------
__global__ __launch_bounds__(256)
void local_agg_kernel(const float* __restrict__ emb,
                      const int* __restrict__ inputs,
                      const int* __restrict__ adj,
                      const float* __restrict__ a_local,
                      float* __restrict__ out) {
  __shared__ float h[S_][D_];      // 25.6 KB
  __shared__ float al[S_][S_];     // 10 KB
  __shared__ float a4[4][D_];      // 2 KB
  const int b = blockIdx.x, tid = threadIdx.x;

  for (int i = tid; i < S_ * D_; i += 256) {
    int s = i >> 7, d = i & 127;
    h[s][d] = emb[(size_t)clampi(inputs[b * S_ + s]) * D_ + d];
  }
  for (int i = tid; i < 4 * D_; i += 256) a4[i >> 7][i & 127] = a_local[i];
  __syncthreads();

  for (int p = tid; p < S_ * S_; p += 256) {
    int i = p / S_, j = p - i * S_;
    int k = adj[((size_t)b * S_ + i) * S_ + j];
    float e = -9e15f;
    if (k >= 1 && k <= 4) {
      float s = 0.f;
      for (int d = 0; d < D_; ++d) s += h[i][d] * h[j][d] * a4[k - 1][d];
      e = leaky02(s);
    }
    al[i][j] = e;
  }
  __syncthreads();

  if (tid < S_) {
    float mx = -3.0e38f;
    for (int j = 0; j < S_; ++j) mx = fmaxf(mx, al[tid][j]);
    float sum = 0.f;
    for (int j = 0; j < S_; ++j) { float e = __expf(al[tid][j] - mx); al[tid][j] = e; sum += e; }
    float inv = 1.f / sum;
    for (int j = 0; j < S_; ++j) al[tid][j] *= inv;
  }
  __syncthreads();

  for (int q = tid; q < S_ * D_; q += 256) {
    int i = q >> 7, d = q & 127;
    float s = 0.f;
    for (int j = 0; j < S_; ++j) s += al[i][j] * h[j][d];
    out[((size_t)b * S_ + i) * D_ + d] = s;
  }
}

// ---------------------------------------------------------------------------
// Global aggregation pass (WMMA). One block = (batch b, tile of TL_ nodes).
// GEMM1: feat[96 x 129] @ w1[129 x 128]  (K=128 on WMMA + rank-1 weight row)
// -> leaky -> fold w2 dot into per-row logits (LDS atomics) -> softmax(12)
// -> neigh = sum_k alpha*neigh_v
// GEMM2: [self | neigh] (8 valid rows of 16) x 256 @ 256x128 -> relu -> out
// ---------------------------------------------------------------------------
__global__ __launch_bounds__(256)
void global_agg_kernel(const float* __restrict__ emb,
                       const float* __restrict__ session,
                       const int* __restrict__ self_idx,     // or null
                       const float* __restrict__ self_dense, // or null
                       const int* __restrict__ neigh_idx,    // or null
                       const float* __restrict__ neigh_dense,// or null
                       const float* __restrict__ neigh_wt,   // [B, L*12]
                       const unsigned short* __restrict__ w1f, // hop slice
                       const float* __restrict__ w1last,       // w1[128,:]
                       const float* __restrict__ w2,           // [128]
                       const unsigned short* __restrict__ w3f, // hop slice
                       float* __restrict__ out, int L, int addToOut) {
  __shared__ float sess[D_];                      // 0.5 KB
  __shared__ unsigned short nvraw[TL_ * SMP_][D_];// 24 KB raw bf16 neighbors
  __shared__ unsigned short featbuf[96 * FP_];    // 25.5 KB feat rows (GEMM1 A),
                                                  // reused as GEMM2 A [16][AP_]
  __shared__ float nw[TL_ * SMP_];
  __shared__ float logit[TL_ * SMP_];
  __shared__ float alpha[TL_ * SMP_];
  __shared__ float selfv[TL_][D_];                // 4 KB
  __shared__ float nsum[TL_][D_];                 // 4 KB

  const int b = blockIdx.y;
  const int l0 = blockIdx.x * TL_;
  const int tid = threadIdx.x;
  const int lane = tid & 31;
  const int wv = tid >> 5;           // wave id 0..7 -> N tile
  const int R = L * SMP_;
  const int RT = TL_ * SMP_;         // 96 rows in this tile

  // phase 0: session, logit init, neighbor weights
  if (tid < D_) sess[tid] = session[b * D_ + tid];
  if (tid < RT) {
    logit[tid] = 0.f;
    int rr = l0 * SMP_ + tid; if (rr > R - 1) rr = R - 1;
    nw[tid] = neigh_wt[(size_t)b * R + rr];
  }
  __syncthreads();

  // phase 1: gather neighbor rows -> raw bf16 + session-scaled feat bf16
  for (int i = tid; i < RT * (D_ / 4); i += 256) {
    int r = i >> 5, c4 = i & 31;
    int rr = l0 * SMP_ + r; if (rr > R - 1) rr = R - 1;
    const float* src;
    if (neigh_idx) src = emb + (size_t)clampi(neigh_idx[(size_t)b * R + rr]) * D_;
    else           src = neigh_dense + ((size_t)b * R + rr) * D_;
    float4 v = ((const float4*)src)[c4];
    float4 sv = ((const float4*)sess)[c4];
    int cc = c4 * 4;
    nvraw[r][cc + 0] = f2bfbits(v.x);
    nvraw[r][cc + 1] = f2bfbits(v.y);
    nvraw[r][cc + 2] = f2bfbits(v.z);
    nvraw[r][cc + 3] = f2bfbits(v.w);
    unsigned short* fr = featbuf + r * FP_ + cc;
    fr[0] = f2bfbits(v.x * sv.x);
    fr[1] = f2bfbits(v.y * sv.y);
    fr[2] = f2bfbits(v.z * sv.z);
    fr[3] = f2bfbits(v.w * sv.w);
  }
  // gather self rows (f32 LDS)
  for (int i = tid; i < TL_ * (D_ / 4); i += 256) {
    int l = i >> 5, c4 = i & 31;
    int ll = l0 + l; if (ll > L - 1) ll = L - 1;
    const float* src;
    if (self_idx) src = emb + (size_t)clampi(self_idx[(size_t)b * L + ll]) * D_;
    else          src = self_dense + ((size_t)b * L + ll) * D_;
    ((float4*)&selfv[l][0])[c4] = ((const float4*)src)[c4];
  }
  __syncthreads();

  // ---- GEMM1: 96x128 @ 128x128, wave wv owns N-tile nt, 6 M-tiles, K=4x32 ----
  const int nt = wv;
  const int hi8 = (lane >> 4) << 3;
  v16bf bfr[4];
#pragma unroll
  for (int ks = 0; ks < 4; ++ks)
    bfr[ks] = *(const v16bf*)(w1f + (((ks * 8 + nt) * 32 + lane) << 4));

  const int nglob = nt * 16 + (lane & 15);
  const float wlastn = w1last[nglob];
  const float w2n = w2[nglob];

  for (int mt = 0; mt < 6; ++mt) {
    v8f acc = {};
    const unsigned short* arow = featbuf + (mt * 16 + (lane & 15)) * FP_;
#pragma unroll
    for (int ks = 0; ks < 4; ++ks) {
      v16bf a = load_afrag(arow, ks * 32, hi8);
      acc = __builtin_amdgcn_wmma_f32_16x16x32_bf16(false, a, false, bfr[ks],
                                                    (short)0, acc, false, false);
    }
#pragma unroll
    for (int j = 0; j < 8; ++j) {
      int mr = mt * 16 + j + hi8;
      float c = acc[j] + nw[mr] * wlastn;       // K=129th row (neigh weight)
      atomicAdd(&logit[mr], leaky02(c) * w2n);  // fold a@w2 without storing a
    }
  }
  __syncthreads();

  // ---- softmax over the 12 samples of each node ----
  if (tid < TL_) {
    int base = tid * SMP_;
    float mx = -3.0e38f;
    for (int k = 0; k < SMP_; ++k) mx = fmaxf(mx, logit[base + k]);
    float s = 0.f;
    for (int k = 0; k < SMP_; ++k) { float e = __expf(logit[base + k] - mx); alpha[base + k] = e; s += e; }
    float inv = 1.f / s;
    for (int k = 0; k < SMP_; ++k) alpha[base + k] *= inv;
  }
  __syncthreads();

  // ---- weighted neighbor sum ----
  for (int i = tid; i < TL_ * D_; i += 256) {
    int l = i >> 7, d = i & 127;
    float s = 0.f;
#pragma unroll
    for (int k = 0; k < SMP_; ++k) s += alpha[l * SMP_ + k] * bf2f(nvraw[l * SMP_ + k][d]);
    nsum[l][d] = s;
  }
  __syncthreads();

  // ---- build GEMM2 A = [self | neigh] bf16 rows (reuse featbuf; GEMM1 done) ----
  unsigned short* a2 = featbuf;
  for (int i = tid; i < 16 * 256; i += 256) {
    int m = i >> 8, k = i & 255;
    float f = 0.f;
    if (m < TL_) f = (k < D_) ? selfv[m][k] : nsum[m][k - D_];
    a2[m * AP_ + k] = f2bfbits(f);
  }
  __syncthreads();

  // ---- GEMM2: 16x256 @ 256x128 (rows 0..7 valid) ----
  {
    v8f acc = {};
    const unsigned short* arow = a2 + (lane & 15) * AP_;
#pragma unroll
    for (int ks = 0; ks < 8; ++ks) {
      v16bf a = load_afrag(arow, ks * 32, hi8);
      v16bf bm = *(const v16bf*)(w3f + (((ks * 8 + nt) * 32 + lane) << 4));
      acc = __builtin_amdgcn_wmma_f32_16x16x32_bf16(false, a, false, bm,
                                                    (short)0, acc, false, false);
    }
    if (lane < 16) {   // rows 0..7 (valid) live in lanes 0-15
#pragma unroll
      for (int j = 0; j < 8; ++j) {
        int l = l0 + j;
        if (l < L) {
          size_t off = ((size_t)b * L + l) * D_ + nglob;
          float v = fmaxf(acc[j], 0.f);
          if (addToOut) out[off] += v; else out[off] = v;
        }
      }
    }
  }
}

// ---------------------------------------------------------------------------
extern "C" void kernel_launch(void* const* d_in, const int* in_sizes, int n_in,
                              void* d_out, int out_size, void* d_ws, size_t ws_size,
                              hipStream_t stream) {
  const int*   inputs     = (const int*)d_in[0];   // [B,S]
  const int*   adj        = (const int*)d_in[1];   // [B,S,S]
  const int*   mask_item  = (const int*)d_in[2];   // [B,S]
  const int*   item       = (const int*)d_in[3];   // [B,S]
  const int*   adj_all    = (const int*)d_in[4];   // [NN,SMP]
  const float* num_weight = (const float*)d_in[5]; // [NN,SMP]
  const float* embedding  = (const float*)d_in[6]; // [NN,D]
  const float* a_local    = (const float*)d_in[7]; // [4,D]
  const float* agg_w1     = (const float*)d_in[8]; // [2,129,128]
  const float* agg_w2     = (const float*)d_in[9]; // [2,128,1]
  const float* agg_w3     = (const float*)d_in[10];// [2,256,128]
  float* out = (float*)d_out;

  // workspace carve-up (all sizes multiples of 256B)
  char* ws = (char*)d_ws;
  size_t off = 0;
  float* session = (float*)(ws + off); off += (size_t)B_ * D_ * 4;                 // 64 KB
  int*   n1      = (int*)  (ws + off); off += (size_t)B_ * S_ * SMP_ * 4;          // 300 KB
  float* nw1     = (float*)(ws + off); off += (size_t)B_ * S_ * SMP_ * 4;
  int*   n2      = (int*)  (ws + off); off += (size_t)B_ * S_ * SMP_ * SMP_ * 4;   // 3.6 MB
  float* nw2     = (float*)(ws + off); off += (size_t)B_ * S_ * SMP_ * SMP_ * 4;
  float* g0      = (float*)(ws + off); off += (size_t)B_ * S_ * D_ * 4;            // 3.3 MB
  float* g1      = (float*)(ws + off); off += (size_t)B_ * S_ * SMP_ * D_ * 4;     // 39 MB
  unsigned short* w1f = (unsigned short*)(ws + off); off += 2 * 4 * 8 * 32 * 16 * 2;
  unsigned short* w3f = (unsigned short*)(ws + off); off += 2 * 8 * 8 * 32 * 16 * 2;
  (void)ws_size; (void)in_sizes; (void)n_in; (void)out_size;

  // 1) weight fragment shuffle (bf16)
  prep_frags_kernel<<<384, 256, 0, stream>>>(agg_w1, agg_w3, w1f, w3f);
  // 2) neighbor sampling tables
  build_n1_kernel<<<(B_ * S_ * SMP_ + 255) / 256, 256, 0, stream>>>(inputs, adj_all, num_weight, n1, nw1);
  build_n2_kernel<<<(B_ * S_ * SMP_ * SMP_ + 255) / 256, 256, 0, stream>>>(n1, adj_all, num_weight, n2, nw2);
  // 3) session embedding
  session_kernel<<<B_, D_, 0, stream>>>(embedding, item, mask_item, session);
  // 4) local aggregation -> d_out = h_local
  local_agg_kernel<<<B_, 256, 0, stream>>>(embedding, inputs, adj, a_local, out);

  const int f1 = 4 * 8 * 32 * 16;   // w1 frag elems per hop
  const int f3 = 8 * 8 * 32 * 16;   // w3 frag elems per hop
  // 5) n_hop=0, hop=0: entity0' = agg(emb[inputs], emb[n1]) -> g0   (L=50)
  global_agg_kernel<<<dim3(7, B_), 256, 0, stream>>>(
      embedding, session, inputs, nullptr, n1, nullptr, nw1,
      w1f, agg_w1 + (0 * 129 + 128) * 128, agg_w2 + 0 * 128, w3f,
      g0, S_, 0);
  // 6) n_hop=0, hop=1: entity1' = agg(emb[n1], emb[n2]) -> g1       (L=600)
  global_agg_kernel<<<dim3(75, B_), 256, 0, stream>>>(
      embedding, session, n1, nullptr, n2, nullptr, nw2,
      w1f, agg_w1 + (0 * 129 + 128) * 128, agg_w2 + 0 * 128, w3f,
      g1, S_ * SMP_, 0);
  // 7) n_hop=1, hop=0: h_global = agg(g0, g1) -> d_out += result    (L=50)
  global_agg_kernel<<<dim3(7, B_), 256, 0, stream>>>(
      embedding, session, nullptr, g0, nullptr, g1, nw1,
      w1f + f1, agg_w1 + (1 * 129 + 128) * 128, agg_w2 + 1 * 128, w3f + f3,
      out, S_, 1);
}
